// DiTBlock_9328668967119
// MI455X (gfx1250) — matrix operand order repaired and checked by reading, not compiled, not run on hardware
//
#include <hip/hip_runtime.h>
#include <hip/hip_bf16.h>

typedef __attribute__((ext_vector_type(16))) _Float16 v16h;
typedef __attribute__((ext_vector_type(8)))  float    v8f;

#define S_TOK   4096
#define DIM     1152
#define D3      3456
#define D6      6912
#define HIDM    4608
#define NEXP    8
#define KCAP    1024
#define NHEAD   16
#define HD      72
#define HDP     96
#define NSEQ    1024
#define NB      4

__device__ inline v8f v8f_zero() {
    v8f z;
#pragma unroll
    for (int i = 0; i < 8; ++i) z[i] = 0.0f;
    return z;
}

// CDNA5 async VMEM->LDS copy (16B per lane), tracked with ASYNCcnt.
__device__ inline void async_b128_to_lds(const void* gsrc, void* ldst) {
    unsigned loff = (unsigned)(unsigned long long)ldst;   // LDS addr = low 32 bits
    asm volatile("global_load_async_to_lds_b128 %0, %1, off"
                 :: "v"(loff), "v"(gsrc) : "memory");
}
__device__ inline void wait_asynccnt0() {
    asm volatile("s_wait_asynccnt 0x0" ::: "memory");
}

// ---------------------------------------------------------------------------
// 1) adaLN: m[b][j] = sum_d silu(c[b][d]) * w_adaln[j][d] + b_adaln[j]
// ---------------------------------------------------------------------------
__global__ __launch_bounds__(256) void k_adaln(const float* __restrict__ c,
                                               const float* __restrict__ w,
                                               const float* __restrict__ bb,
                                               float* __restrict__ m) {
    int o = blockIdx.x * 256 + threadIdx.x;        // 0 .. 4*6912-1
    if (o >= NB * D6) return;
    int b = o / D6, j = o % D6;
    const float* cr = c + (size_t)b * DIM;
    const float* wr = w + (size_t)j * DIM;
    float acc = bb[j];
    for (int d = 0; d < DIM; ++d) {
        float cv = cr[d];
        cv = cv / (1.0f + __expf(-cv));            // silu
        acc += cv * wr[d];
    }
    m[o] = acc;
}

// ---------------------------------------------------------------------------
// 2) LayerNorm + modulate -> f16
// ---------------------------------------------------------------------------
__global__ __launch_bounds__(256) void k_ln_mod(const float* __restrict__ x,
                                                const float* __restrict__ m,
                                                _Float16* __restrict__ hout,
                                                int shift_sec, int scale_sec) {
    int token = blockIdx.x;                        // 0..4095
    int b = token >> 10;
    const float* xr = x + (size_t)token * DIM;
    __shared__ float red[256];
    float s = 0.f, s2 = 0.f;
    for (int i = threadIdx.x; i < DIM; i += 256) {
        float v = xr[i]; s += v; s2 += v * v;
    }
    red[threadIdx.x] = s; __syncthreads();
    for (int st = 128; st > 0; st >>= 1) {
        if (threadIdx.x < st) red[threadIdx.x] += red[threadIdx.x + st];
        __syncthreads();
    }
    float tot = red[0]; __syncthreads();
    red[threadIdx.x] = s2; __syncthreads();
    for (int st = 128; st > 0; st >>= 1) {
        if (threadIdx.x < st) red[threadIdx.x] += red[threadIdx.x + st];
        __syncthreads();
    }
    float tot2 = red[0];
    float mu = tot / (float)DIM;
    float var = tot2 / (float)DIM - mu * mu;
    float rstd = rsqrtf(var + 1e-6f);
    const float* mb = m + (size_t)b * D6;
    for (int i = threadIdx.x; i < DIM; i += 256) {
        float sc = mb[scale_sec * DIM + i];
        float sh = mb[shift_sec * DIM + i];
        float v = (xr[i] - mu) * rstd * (1.0f + sc) + sh;
        hout[(size_t)token * DIM + i] = (_Float16)v;
    }
}

// ---------------------------------------------------------------------------
// 3) WMMA GEMM: Out[bz] = A[bz](f16, MxK) * B[bz](f32, NxK)^T + bias
//    128x64 block tile, 256 threads (8 waves), v_wmma_f32_16x16x32_f16.
//    A tiles DMA'd into LDS via global_load_async_to_lds_b128 (ASYNCcnt);
//    B tiles converted f32->f16 through VGPRs.
//    EPI: 0=identity 1=silu 2=gelu(tanh).  OUT_HALF: f16 or f32 output.
// ---------------------------------------------------------------------------
template <int EPI, bool OUT_HALF>
__global__ __launch_bounds__(256) void k_gemm_wmma(
    const _Float16* __restrict__ A, const float* __restrict__ Bw,
    const float* __restrict__ bias, void* __restrict__ Out,
    int M, int Nn, int K) {
    __shared__ _Float16 As[128][32];
    __shared__ _Float16 Bs[64][32];
    const int bz = blockIdx.z;
    const int m0 = blockIdx.y * 128;
    const int n0 = blockIdx.x * 64;
    A    += (size_t)bz * M * K;
    Bw   += (size_t)bz * Nn * K;
    bias += (size_t)bz * Nn;
    const int tid  = threadIdx.x;
    const int wave = tid >> 5;            // 0..7, owns rows m0+wave*16..+16
    const int lane = tid & 31;
    const int half = lane >> 4;
    const int l16  = lane & 15;
    const int arow  = tid >> 1;           // 0..127  A staged row
    const int ahalf = tid & 1;            // 16-half of the K chunk
    const int brow  = tid >> 2;           // 0..63   B staged row
    const int bq    = tid & 3;            // 8-float quarter of the K chunk

    v8f acc[4];
#pragma unroll
    for (int i = 0; i < 4; ++i) acc[i] = v8f_zero();

    for (int kc = 0; kc < K; kc += 32) {
        __syncthreads();
        // A stage: async DMA straight to LDS (2 x 16B per thread)
        {
            const _Float16* ag = A + (size_t)(m0 + arow) * K + kc + ahalf * 16;
            async_b128_to_lds(ag,     &As[arow][ahalf * 16]);
            async_b128_to_lds(ag + 8, &As[arow][ahalf * 16 + 8]);
        }
        // B stage: f32 -> f16 convert (8 floats -> 16B per thread)
        {
            const float4* bg = (const float4*)(Bw + (size_t)(n0 + brow) * K + kc + bq * 8);
            float4 b0 = bg[0], b1 = bg[1];
            _Float16* bd = &Bs[brow][bq * 8];
            bd[0]=(_Float16)b0.x; bd[1]=(_Float16)b0.y; bd[2]=(_Float16)b0.z; bd[3]=(_Float16)b0.w;
            bd[4]=(_Float16)b1.x; bd[5]=(_Float16)b1.y; bd[6]=(_Float16)b1.z; bd[7]=(_Float16)b1.w;
        }
        wait_asynccnt0();
        __syncthreads();
        // A fragment (contiguous K: lane half selects K 16-window, matching B)
        v16h af;
        {
            const uint4* p0 = (const uint4*)&As[wave * 16 + l16][half * 16];
            const uint4* p1 = (const uint4*)&As[wave * 16 + l16][half * 16 + 8];
            ((uint4*)&af)[0] = *p0;
            ((uint4*)&af)[1] = *p1;
        }
#pragma unroll
        for (int nt = 0; nt < 4; ++nt) {
            v16h bf;
            const uint4* q = (const uint4*)&Bs[nt * 16 + l16][half * 16];
            ((uint4*)&bf)[0] = q[0];
            ((uint4*)&bf)[1] = q[1];
            acc[nt] = __builtin_amdgcn_wmma_f32_16x16x32_f16(
                false, af, false, bf, (short)0, acc[nt], false, false);
        }
    }

    // Epilogue: C layout lane l -> row r + 8*half, col l16
    const int orow = m0 + wave * 16 + half * 8;
#pragma unroll
    for (int nt = 0; nt < 4; ++nt) {
        int col = n0 + nt * 16 + l16;
        float bv = bias[col];
#pragma unroll
        for (int r = 0; r < 8; ++r) {
            float v = acc[nt][r] + bv;
            if (EPI == 1) {
                v = v / (1.0f + __expf(-v));                       // silu
            } else if (EPI == 2) {
                float u = 0.7978845608028654f * (v + 0.044715f * v * v * v);
                v = 0.5f * v * (1.0f + tanhf(u));                  // gelu tanh
            }
            size_t off = (size_t)bz * M * Nn + (size_t)(orow + r) * Nn + col;
            if (OUT_HALF) ((_Float16*)Out)[off] = (_Float16)v;
            else          ((float*)Out)[off]   = v;
        }
    }
}

// ---------------------------------------------------------------------------
// 4) repack qkv(f32 [S][3][16][72]) -> qf/kf [bh][n][96] f16 (zero-padded),
//    vT [bh][96][1024] f16
// ---------------------------------------------------------------------------
__global__ __launch_bounds__(256) void k_repack_qkv(const float* __restrict__ qkv,
                                                    _Float16* __restrict__ qf,
                                                    _Float16* __restrict__ kf,
                                                    _Float16* __restrict__ vt) {
    size_t i = (size_t)blockIdx.x * 256 + threadIdx.x;   // over B*NH*N*96
    if (i >= (size_t)NB * NHEAD * NSEQ * HDP) return;
    int d = (int)(i % HDP);
    size_t t = i / HDP;
    int n = (int)(t % NSEQ);
    size_t bh = t / NSEQ;
    int hh = (int)(bh % NHEAD);
    int b  = (int)(bh / NHEAD);
    size_t tok = (size_t)b * NSEQ + n;
    float qv = 0.f, kv = 0.f, vv = 0.f;
    if (d < HD) {
        const float* base = qkv + tok * D3;
        qv = base[0 * DIM + hh * HD + d];
        kv = base[1 * DIM + hh * HD + d];
        vv = base[2 * DIM + hh * HD + d];
    }
    qf[i] = (_Float16)qv;
    kf[i] = (_Float16)kv;
    vt[(bh * HDP + d) * NSEQ + n] = (_Float16)vv;
}

// ---------------------------------------------------------------------------
// 5) flash attention: 1 wave = 16 query rows; 32 keys/iter (2 S-tiles),
//    6 PV WMMAs into 16x96 f32 accumulator. o -> f16 [b][n][16*72]
// ---------------------------------------------------------------------------
__global__ __launch_bounds__(128) void k_attn(const _Float16* __restrict__ qf,
                                              const _Float16* __restrict__ kf,
                                              const _Float16* __restrict__ vt,
                                              _Float16* __restrict__ o) {
    const int wave = threadIdx.x >> 5;
    const int lane = threadIdx.x & 31;
    const int half = lane >> 4;
    const int l16  = lane & 15;
    const int qt = blockIdx.x * 4 + wave;     // 0..63
    const int h  = blockIdx.y;
    const int b  = blockIdx.z;
    const size_t bh = (size_t)(b * NHEAD + h);
    const _Float16* qb = qf + bh * NSEQ * HDP;
    const _Float16* kb = kf + bh * NSEQ * HDP;
    const _Float16* vb = vt + bh * HDP * NSEQ;

    __shared__ _Float16 Pl[4][16][32];

    // preload 3 Q A-fragments (head dim padded to 96 = 3 x 32)
    v16h qfr[3];
    const int qrow = qt * 16 + l16;
#pragma unroll
    for (int c = 0; c < 3; ++c) {
        const uint4* p0 = (const uint4*)(qb + (size_t)qrow * HDP + c * 32 + half * 16);
        const uint4* p1 = (const uint4*)(qb + (size_t)qrow * HDP + c * 32 + half * 16 + 8);
        ((uint4*)&qfr[c])[0] = *p0;
        ((uint4*)&qfr[c])[1] = *p1;
    }

    v8f accO[6];
#pragma unroll
    for (int t = 0; t < 6; ++t) accO[t] = v8f_zero();
    float mi[8], li[8];
#pragma unroll
    for (int r = 0; r < 8; ++r) { mi[r] = -1e30f; li[r] = 0.f; }
    const float scale = rsqrtf((float)HD);

    for (int kb32 = 0; kb32 < NSEQ / 32; ++kb32) {
        v8f Sv[2];
#pragma unroll
        for (int s = 0; s < 2; ++s) {
            v8f sa = v8f_zero();
#pragma unroll
            for (int c = 0; c < 3; ++c) {
                v16h kfr;
                const _Float16* kp = kb + (size_t)(kb32 * 32 + s * 16 + l16) * HDP
                                        + c * 32 + half * 16;
                ((uint4*)&kfr)[0] = ((const uint4*)kp)[0];
                ((uint4*)&kfr)[1] = ((const uint4*)kp)[1];
                sa = __builtin_amdgcn_wmma_f32_16x16x32_f16(
                    false, qfr[c], false, kfr, (short)0, sa, false, false);
            }
            Sv[s] = sa;
        }
        // online softmax (rows live across the 16 lanes of each half)
        float fac[8];
#pragma unroll
        for (int r = 0; r < 8; ++r) {
            float v0 = Sv[0][r] * scale;
            float v1 = Sv[1][r] * scale;
            float mx = fmaxf(v0, v1);
#pragma unroll
            for (int off = 1; off < 16; off <<= 1)
                mx = fmaxf(mx, __shfl_xor(mx, off, 32));
            float mnew = fmaxf(mi[r], mx);
            float p0 = __expf(v0 - mnew);
            float p1 = __expf(v1 - mnew);
            float rs = p0 + p1;
#pragma unroll
            for (int off = 1; off < 16; off <<= 1)
                rs += __shfl_xor(rs, off, 32);
            fac[r] = __expf(mi[r] - mnew);
            li[r] = li[r] * fac[r] + rs;
            mi[r] = mnew;
            Sv[0][r] = p0; Sv[1][r] = p1;
        }
#pragma unroll
        for (int t = 0; t < 6; ++t)
#pragma unroll
            for (int r = 0; r < 8; ++r) accO[t][r] *= fac[r];

        // transpose P (C layout) -> A layout via per-wave LDS bounce
        __syncthreads();
#pragma unroll
        for (int s = 0; s < 2; ++s)
#pragma unroll
            for (int r = 0; r < 8; ++r)
                Pl[wave][r + 8 * half][s * 16 + l16] = (_Float16)Sv[s][r];
        __syncthreads();
        v16h pa;
        {
            const uint4* p0 = (const uint4*)&Pl[wave][l16][half * 16];
            const uint4* p1 = (const uint4*)&Pl[wave][l16][half * 16 + 8];
            ((uint4*)&pa)[0] = *p0;
            ((uint4*)&pa)[1] = *p1;
        }
        // PV: 6 d-tiles of 16, contraction over the 32 keys
#pragma unroll
        for (int t = 0; t < 6; ++t) {
            v16h vfr;
            const _Float16* vp = vb + (size_t)(t * 16 + l16) * NSEQ + kb32 * 32 + half * 16;
            ((uint4*)&vfr)[0] = ((const uint4*)vp)[0];
            ((uint4*)&vfr)[1] = ((const uint4*)vp)[1];
            accO[t] = __builtin_amdgcn_wmma_f32_16x16x32_f16(
                false, pa, false, vfr, (short)0, accO[t], false, false);
        }
    }

    // finalize: divide by l, store d<72 columns into o[b][n][h*72+d] (f16)
#pragma unroll
    for (int r = 0; r < 8; ++r) {
        float inv = 1.0f / li[r];
        int n = qt * 16 + r + 8 * half;
        size_t rowbase = ((size_t)b * NSEQ + n) * DIM + h * HD;
#pragma unroll
        for (int t = 0; t < 5; ++t) {
            int d = t * 16 + l16;
            if (d < HD) o[rowbase + d] = (_Float16)(accO[t][r] * inv);
        }
    }
}

// ---------------------------------------------------------------------------
// 6) residuals
// ---------------------------------------------------------------------------
__global__ __launch_bounds__(256) void k_residual(const float* __restrict__ xin,
                                                  const float* __restrict__ add,
                                                  const float* __restrict__ m,
                                                  int gate_sec,
                                                  float* __restrict__ xout) {
    size_t i = (size_t)blockIdx.x * 256 + threadIdx.x;
    if (i >= (size_t)S_TOK * DIM) return;
    int col = (int)(i % DIM);
    int b = (int)(i / ((size_t)NSEQ * DIM));
    float g = m[(size_t)b * D6 + gate_sec * DIM + col];
    xout[i] = xin[i] + g * add[i];
}

// ---------------------------------------------------------------------------
// 7) gate weights: wbar[e][d] = mean_g gates_w[g][e][d]; bbar[e] = mean gates_b
// ---------------------------------------------------------------------------
__global__ __launch_bounds__(256) void k_wbar(const float* __restrict__ gw,
                                              const float* __restrict__ gb,
                                              float* __restrict__ wbar,
                                              float* __restrict__ bbar) {
    int i = blockIdx.x * 256 + threadIdx.x;
    if (i < NEXP * DIM) {
        float s = 0.f;
        for (int g = 0; g < 4; ++g) s += gw[(size_t)g * NEXP * DIM + i];
        wbar[i] = s * 0.25f;
    }
    if (i < NEXP) {
        float s = 0.f;
        for (int g = 0; g < 4; ++g) s += gb[g * NEXP + i];
        bbar[i] = s * 0.25f;
    }
}

// 8) router scores: softmax_e( h2[s] . wbar[e] + bbar[e] ) -> scores[e][s]
__global__ __launch_bounds__(256) void k_scores(const _Float16* __restrict__ h2,
                                                const float* __restrict__ wbar,
                                                const float* __restrict__ bbar,
                                                float* __restrict__ scores) {
    int s = blockIdx.x * 256 + threadIdx.x;
    if (s >= S_TOK) return;
    float lg[NEXP];
#pragma unroll
    for (int e = 0; e < NEXP; ++e) lg[e] = bbar[e];
    const _Float16* hr = h2 + (size_t)s * DIM;
    for (int d = 0; d < DIM; ++d) {
        float hv = (float)hr[d];
#pragma unroll
        for (int e = 0; e < NEXP; ++e) lg[e] += hv * wbar[e * DIM + d];
    }
    float mx = lg[0];
#pragma unroll
    for (int e = 1; e < NEXP; ++e) mx = fmaxf(mx, lg[e]);
    float sum = 0.f;
#pragma unroll
    for (int e = 0; e < NEXP; ++e) { lg[e] = __expf(lg[e] - mx); sum += lg[e]; }
    float inv = 1.0f / sum;
#pragma unroll
    for (int e = 0; e < NEXP; ++e) scores[(size_t)e * S_TOK + s] = lg[e] * inv;
}

// 9) per-expert exact top-1024 of 4096 via LDS bitonic sort (desc, idx tiebreak)
__global__ __launch_bounds__(1024) void k_topk(const float* __restrict__ scores,
                                               int* __restrict__ idx,
                                               float* __restrict__ gv,
                                               float* __restrict__ ones) {
    __shared__ unsigned long long key[S_TOK];
    int e = blockIdx.x;
    int tid = threadIdx.x;
    for (int i = tid; i < S_TOK; i += 1024) {
        float v = scores[(size_t)e * S_TOK + i];
        unsigned u = __float_as_uint(v);
        u = (u & 0x80000000u) ? ~u : (u | 0x80000000u);   // order-preserving
        key[i] = ((unsigned long long)u << 32) | (unsigned)(0xFFFFFFFFu - i);
    }
    __syncthreads();
    for (unsigned size = 2; size <= S_TOK; size <<= 1) {
        for (unsigned stride = size >> 1; stride > 0; stride >>= 1) {
            for (unsigned t = tid; t < S_TOK / 2; t += 1024) {
                unsigned i = ((t / stride) * stride * 2) + (t % stride);
                unsigned j = i + stride;
                bool desc = ((i & size) == 0);
                unsigned long long a = key[i], b2 = key[j];
                bool doswap = desc ? (a < b2) : (a > b2);
                if (doswap) { key[i] = b2; key[j] = a; }
            }
            __syncthreads();
        }
    }
    if (tid < KCAP) {
        unsigned long long kk = key[tid];
        unsigned s = 0xFFFFFFFFu - (unsigned)(kk & 0xFFFFFFFFu);
        unsigned u = (unsigned)(kk >> 32);
        u = (u & 0x80000000u) ? (u & 0x7FFFFFFFu) : ~u;
        idx[e * KCAP + tid] = (int)s;
        gv[e * KCAP + tid] = __uint_as_float(u);
        ones[(size_t)s * NEXP + e] = 1.0f;
    }
}

// 10) gather tokens per expert (f16)
__global__ __launch_bounds__(256) void k_gather(const _Float16* __restrict__ h2,
                                                const int* __restrict__ idx,
                                                _Float16* __restrict__ expin) {
    size_t i = (size_t)blockIdx.x * 256 + threadIdx.x;
    if (i >= (size_t)NEXP * KCAP * DIM) return;
    int d = (int)(i % DIM);
    size_t t = i / DIM;
    int r = (int)(t % KCAP);
    int e = (int)(t / KCAP);
    expin[i] = h2[(size_t)idx[e * KCAP + r] * DIM + d];
}

// 11) gated scatter-add of expert outputs
__global__ __launch_bounds__(256) void k_scatter(const float* __restrict__ moeraw,
                                                 const int* __restrict__ idx,
                                                 const float* __restrict__ gv,
                                                 float* __restrict__ y) {
    size_t i = (size_t)blockIdx.x * 256 + threadIdx.x;
    if (i >= (size_t)NEXP * KCAP * DIM) return;
    int d = (int)(i % DIM);
    size_t t = i / DIM;
    int r = (int)(t % KCAP);
    int e = (int)(t / KCAP);
    atomicAdd(&y[(size_t)idx[e * KCAP + r] * DIM + d], gv[e * KCAP + r] * moeraw[i]);
}

// 12) capacity head layer 2 (N=8 too small for WMMA)
__global__ __launch_bounds__(256) void k_cap2(const _Float16* __restrict__ caph,
                                              const float* __restrict__ w2,
                                              const float* __restrict__ b2,
                                              float* __restrict__ out) {
    int i = blockIdx.x * 256 + threadIdx.x;      // over S*E
    if (i >= S_TOK * NEXP) return;
    int s = i / NEXP, e = i % NEXP;
    float acc = b2[e];
    const _Float16* hr = caph + (size_t)s * DIM;
    const float* wr = w2 + (size_t)e * DIM;
    for (int d = 0; d < DIM; ++d) acc += (float)hr[d] * wr[d];
    out[i] = acc;
}

// ===========================================================================
extern "C" void kernel_launch(void* const* d_in, const int* in_sizes, int n_in,
                              void* d_out, int out_size, void* d_ws, size_t ws_size,
                              hipStream_t stream) {
    const float* x       = (const float*)d_in[0];
    const float* c       = (const float*)d_in[1];
    const float* w_adaln = (const float*)d_in[2];
    const float* b_adaln = (const float*)d_in[3];
    const float* w_qkv   = (const float*)d_in[4];
    const float* b_qkv   = (const float*)d_in[5];
    const float* w_proj  = (const float*)d_in[6];
    const float* b_proj  = (const float*)d_in[7];
    const float* gates_w = (const float*)d_in[8];
    const float* gates_b = (const float*)d_in[9];
    const float* cap_w1  = (const float*)d_in[10];
    const float* cap_b1  = (const float*)d_in[11];
    const float* cap_w2  = (const float*)d_in[12];
    const float* cap_b2  = (const float*)d_in[13];
    const float* exp_w1  = (const float*)d_in[14];
    const float* exp_b1  = (const float*)d_in[15];
    const float* exp_w2  = (const float*)d_in[16];
    const float* exp_b2  = (const float*)d_in[17];

    char* ws = (char*)d_ws;
    size_t off = 0;
    auto alloc = [&](size_t bytes) { size_t o = off; off += (bytes + 255) & ~(size_t)255; return o; };
    const size_t off_m      = alloc((size_t)NB * D6 * 4);
    const size_t off_h      = alloc((size_t)S_TOK * DIM * 2);          // f16
    const size_t off_qkv    = alloc((size_t)S_TOK * D3 * 4);           // f32 (reused for moeraw)
    const size_t off_qf     = alloc((size_t)NB * NHEAD * NSEQ * HDP * 2);
    const size_t off_kf     = alloc((size_t)NB * NHEAD * NSEQ * HDP * 2);
    const size_t off_vt     = alloc((size_t)NB * NHEAD * HDP * NSEQ * 2);
    const size_t off_o      = alloc((size_t)S_TOK * DIM * 2);          // f16
    const size_t off_proj   = alloc((size_t)S_TOK * DIM * 4);
    const size_t off_x1     = alloc((size_t)S_TOK * DIM * 4);
    const size_t off_h2     = alloc((size_t)S_TOK * DIM * 2);
    const size_t off_caph   = alloc((size_t)S_TOK * DIM * 2);
    const size_t off_wbar   = alloc((size_t)NEXP * DIM * 4);
    const size_t off_bbar   = alloc((size_t)NEXP * 4);
    const size_t off_scores = alloc((size_t)NEXP * S_TOK * 4);
    const size_t off_idx    = alloc((size_t)NEXP * KCAP * 4);
    const size_t off_gv     = alloc((size_t)NEXP * KCAP * 4);
    const size_t off_expin  = alloc((size_t)NEXP * KCAP * DIM * 2);
    const size_t off_h1     = alloc((size_t)NEXP * KCAP * HIDM * 2);
    const size_t off_y      = alloc((size_t)S_TOK * DIM * 4);
    const size_t off_moeraw = off_qkv;   // alias: qkv dead after repack
    (void)ws_size; (void)in_sizes; (void)n_in; (void)out_size;

    float*     m      = (float*)(ws + off_m);
    _Float16*  hmsa   = (_Float16*)(ws + off_h);
    float*     qkv    = (float*)(ws + off_qkv);
    _Float16*  qf     = (_Float16*)(ws + off_qf);
    _Float16*  kf     = (_Float16*)(ws + off_kf);
    _Float16*  vt     = (_Float16*)(ws + off_vt);
    _Float16*  obuf   = (_Float16*)(ws + off_o);
    float*     projo  = (float*)(ws + off_proj);
    float*     x1     = (float*)(ws + off_x1);
    _Float16*  h2     = (_Float16*)(ws + off_h2);
    _Float16*  caph   = (_Float16*)(ws + off_caph);
    float*     wbar   = (float*)(ws + off_wbar);
    float*     bbar   = (float*)(ws + off_bbar);
    float*     scores = (float*)(ws + off_scores);
    int*       idx    = (int*)(ws + off_idx);
    float*     gv     = (float*)(ws + off_gv);
    _Float16*  expin  = (_Float16*)(ws + off_expin);
    _Float16*  h1     = (_Float16*)(ws + off_h1);
    float*     ybuf   = (float*)(ws + off_y);
    float*     moeraw = (float*)(ws + off_moeraw);

    float* out_x    = (float*)d_out;                       // [B,N,1152]
    float* out_ones = out_x + (size_t)S_TOK * DIM;         // [B,N,8]
    float* out_cap  = out_ones + (size_t)S_TOK * NEXP;     // [B,N,8]

    // 1) adaLN modulation vector
    k_adaln<<<(NB * D6 + 255) / 256, 256, 0, stream>>>(c, w_adaln, b_adaln, m);
    // 2) h = modulate(ln(x), shift_msa(0), scale_msa(1)) -> f16
    k_ln_mod<<<S_TOK, 256, 0, stream>>>(x, m, hmsa, 0, 1);
    // 3) qkv = h @ w_qkv^T + b  (WMMA, async-LDS staging)
    k_gemm_wmma<0, false><<<dim3(D3 / 64, S_TOK / 128, 1), 256, 0, stream>>>(
        hmsa, w_qkv, b_qkv, qkv, S_TOK, D3, DIM);
    // 4) repack to f16 padded q/k + transposed v
    {
        size_t tot = (size_t)NB * NHEAD * NSEQ * HDP;
        k_repack_qkv<<<(unsigned)((tot + 255) / 256), 256, 0, stream>>>(qkv, qf, kf, vt);
    }
    // 5) flash attention (WMMA) -> o f16
    k_attn<<<dim3(NSEQ / 64, NHEAD, NB), 128, 0, stream>>>(qf, kf, vt, obuf);
    // 6) proj (WMMA)
    k_gemm_wmma<0, false><<<dim3(DIM / 64, S_TOK / 128, 1), 256, 0, stream>>>(
        obuf, w_proj, b_proj, projo, S_TOK, DIM, DIM);
    // 7) x1 = x + gate_msa(2) * proj
    k_residual<<<(S_TOK * DIM + 255) / 256, 256, 0, stream>>>(x, projo, m, 2, x1);
    // 8) h2 = modulate(ln(x1), shift_mlp(3), scale_mlp(4))
    k_ln_mod<<<S_TOK, 256, 0, stream>>>(x1, m, h2, 3, 4);
    // 9) capacity head layer1 (WMMA, silu) -> f16
    k_gemm_wmma<1, true><<<dim3(DIM / 64, S_TOK / 128, 1), 256, 0, stream>>>(
        h2, cap_w1, cap_b1, caph, S_TOK, DIM, DIM);
    // 10) capacity head layer2 -> out_cap
    k_cap2<<<(S_TOK * NEXP + 255) / 256, 256, 0, stream>>>(caph, cap_w2, cap_b2, out_cap);
    // 11-12) router
    k_wbar<<<(NEXP * DIM + 255) / 256, 256, 0, stream>>>(gates_w, gates_b, wbar, bbar);
    k_scores<<<(S_TOK + 255) / 256, 256, 0, stream>>>(h2, wbar, bbar, scores);
    // 13) per-expert top-k (+ones one-hot)
    hipMemsetAsync(out_ones, 0, (size_t)S_TOK * NEXP * 4, stream);
    k_topk<<<NEXP, 1024, 0, stream>>>(scores, idx, gv, out_ones);
    // 14) gather expert inputs
    {
        size_t tot = (size_t)NEXP * KCAP * DIM;
        k_gather<<<(unsigned)((tot + 255) / 256), 256, 0, stream>>>(h2, idx, expin);
    }
    // 15) expert MLP layer1 (WMMA, gelu) -> f16, batched over experts
    k_gemm_wmma<2, true><<<dim3(HIDM / 64, KCAP / 128, NEXP), 256, 0, stream>>>(
        expin, exp_w1, exp_b1, h1, KCAP, HIDM, DIM);
    // 16) expert MLP layer2 (WMMA) -> f32
    k_gemm_wmma<0, false><<<dim3(DIM / 64, KCAP / 128, NEXP), 256, 0, stream>>>(
        h1, exp_w2, exp_b2, moeraw, KCAP, DIM, HIDM);
    // 17) gated scatter-add
    hipMemsetAsync(ybuf, 0, (size_t)S_TOK * DIM * 4, stream);
    {
        size_t tot = (size_t)NEXP * KCAP * DIM;
        k_scatter<<<(unsigned)((tot + 255) / 256), 256, 0, stream>>>(moeraw, idx, gv, ybuf);
    }
    // 18) final residual -> out_x
    k_residual<<<(S_TOK * DIM + 255) / 256, 256, 0, stream>>>(x1, ybuf, m, 5, out_x);
}
